// BidirectionalMambaBlock_50371376448081
// MI455X (gfx1250) — matrix operand order repaired and checked by reading, not compiled, not run on hardware
//
#include <hip/hip_runtime.h>
#include <hip/hip_bf16.h>
#include <math.h>

// ---------------------------------------------------------------------------
// BidirectionalMambaBlock for MI455X (gfx1250, wave32, WMMA).
//   B=8, L=256, d_model=512, d_inner=1024, N(state)=16, dt_rank=32, HID=1024
// GEMMs: v_wmma_f32_16x16x32_f16, 64x64 workgroup tile (4 waves x 16x64),
//        f16 LDS staging, double-buffered K pipeline, all-B-fragments
//        preloaded ahead of the 4-WMMA chain, global prefetch of chunk c+2.
// Scan / activations / LayerNorm in f32.
// ---------------------------------------------------------------------------

#define LTOK   256
#define NBATCH 8
#define DMODEL 512
#define DINNER 1024
#define DSTATE 16
#define DTRANK 32
#define HIDDIM 1024
#define NROWS  (NBATCH * LTOK)   // 2048 token rows

typedef __attribute__((ext_vector_type(16))) _Float16 v16h;
typedef __attribute__((ext_vector_type(2)))  _Float16 v2h;
typedef __attribute__((ext_vector_type(8)))  float    v8f;

__device__ __forceinline__ float silu_f(float v) { return v / (1.0f + expf(-v)); }

// Stage a 64x32 f32 tile as f16 into LDS. 128 threads; thread t covers
// col = t&31, rows (t>>5) + 4*i. Coalesced 32-float rows, one cvt per value.
__device__ __forceinline__ void stage_tile_f16(_Float16 (*__restrict__ dst)[34],
                                               const float* __restrict__ src,
                                               int row0, int ld, int k0, int tid)
{
    const int col = tid & 31;
    const int rb  = tid >> 5;     // 0..3
#pragma unroll
    for (int i = 0; i < 16; ++i) {
        const int r = rb + 4 * i;
        dst[r][col] = (_Float16)src[(size_t)(row0 + r) * ld + k0 + col];
    }
}

// ---------------------------------------------------------------------------
// Generic GEMM:  O[M,N] = A[M,K] * W[N,K]^T + bias  (act: 0=none,1=relu,2=softplus)
// grid = (N/64, M/64), block = 128 (4 waves). Wave w -> rows m0+16w..+15,
// all 64 block columns (4 WMMAs / K-chunk, shared B tile).
// K must be a multiple of 32 (call sites: 512, 1024, 32).
// Fragment layouts per CDNA5 ISA §7.12.2 (16-bit A 16x32, B 32x16);
// K-pairs are contiguous, so gathers are packed 32-bit LDS reads.
// ---------------------------------------------------------------------------
__global__ void gemm_wmma_kernel(const float* __restrict__ A, int lda,
                                 const float* __restrict__ W, int ldw,
                                 const float* __restrict__ bias,
                                 float* __restrict__ O, int ldo,
                                 int K, int act)
{
    __shared__ _Float16 sA[2][64][34];   // [buf][block row][K-chunk]
    __shared__ _Float16 sB[2][64][34];   // [buf][block col][K-chunk]

    const int tid  = threadIdx.x;        // 0..127
    const int lane = tid & 31;
    const int wid  = tid >> 5;           // wave id 0..3
    const int m0   = blockIdx.y * 64;
    const int n0   = blockIdx.x * 64;
    const int ml   = lane & 15;
    const int half = lane >> 4;
    const int arow = wid * 16 + ml;      // A fragment row within block tile

    v8f acc[4] = {};
    const int nchunks = K >> 5;

    // Prologue: stage chunk 0 into buffer 0.
    stage_tile_f16(sA[0], A, m0, lda, 0, tid);
    stage_tile_f16(sB[0], W, n0, ldw, 0, tid);
    __syncthreads();

    int cur = 0;
    for (int c = 0; c < nchunks; ++c) {
        // Pipeline: stage chunk c+1 into the other buffer while computing c,
        // and prefetch chunk c+2 toward cache (gfx1250 global_prefetch_b8;
        // one line per thread covers all 64 A rows + 64 W rows).
        if (c + 1 < nchunks) {
            const int kn = (c + 1) << 5;
            stage_tile_f16(sA[cur ^ 1], A, m0, lda, kn, tid);
            stage_tile_f16(sB[cur ^ 1], W, n0, ldw, kn, tid);
            if (c + 2 < nchunks) {
                const int kp = (c + 2) << 5;
                if (tid < 64) __builtin_prefetch(&A[(size_t)(m0 + tid) * lda + kp], 0, 1);
                else          __builtin_prefetch(&W[(size_t)(n0 + tid - 64) * ldw + kp], 0, 1);
            }
        }

        // A fragment (16-bit A 16x32): pair p -> K = (p<4 ? 2p : 16+2(p-4)) + 8*half.
        v16h af;
#pragma unroll
        for (int p = 0; p < 8; ++p) {
            const int ka = (p < 4 ? 2 * p : 16 + 2 * (p - 4)) + 8 * half;
            const v2h pr = *(const v2h*)&sA[cur][arow][ka];
            af[2 * p]     = pr[0];
            af[2 * p + 1] = pr[1];
        }

        // Preload ALL four B fragments first so the ds loads overlap and the
        // four WMMAs can issue back-to-back (staggered dscnt waits instead of
        // a full s_wait_dscnt 0 stall before every WMMA).
        v16h bf[4];
#pragma unroll
        for (int j = 0; j < 4; ++j) {
#pragma unroll
            for (int p = 0; p < 8; ++p) {
                const int kb = (p < 4 ? 2 * p : 16 + 2 * (p - 4)) + 8 * half;
                const v2h pr = *(const v2h*)&sB[cur][j * 16 + ml][kb];
                bf[j][2 * p]     = pr[0];
                bf[j][2 * p + 1] = pr[1];
            }
        }

#pragma unroll
        for (int j = 0; j < 4; ++j) {
            acc[j] = __builtin_amdgcn_wmma_f32_16x16x32_f16(
                /*neg_a=*/false, af, /*neg_b=*/false, bf[j],
                /*c_mod=*/(short)0, acc[j], /*reuse_a=*/false, /*reuse_b=*/false);
        }

        __syncthreads();
        cur ^= 1;
    }

    // Epilogue. C/D layout: VGPR r holds row (r + 8*half), column = lane&15.
#pragma unroll
    for (int j = 0; j < 4; ++j) {
        const int   col = n0 + j * 16 + ml;
        const float bv  = bias ? bias[col] : 0.0f;
#pragma unroll
        for (int r = 0; r < 8; ++r) {
            const int row = m0 + wid * 16 + r + 8 * half;
            float v = acc[j][r] + bv;
            if (act == 1)      v = fmaxf(v, 0.0f);
            else if (act == 2) v = (v > 20.0f) ? v : log1pf(expf(v));   // softplus
            O[(size_t)row * ldo + col] = v;
        }
    }
}

// ---------------------------------------------------------------------------
// Causal depthwise conv (k=2) + SiLU:  xc = silu(w0*x[t-1] + w1*x[t] + b)
// xz layout: [NROWS, 2*DINNER]; xi = first DINNER cols.
// ---------------------------------------------------------------------------
__global__ void conv_silu_kernel(const float* __restrict__ xz,
                                 const float* __restrict__ cw,   // [DINNER,2]
                                 const float* __restrict__ cb,   // [DINNER]
                                 float* __restrict__ xc)         // [NROWS,DINNER]
{
    const int idx = blockIdx.x * blockDim.x + threadIdx.x;
    if (idx >= NROWS * DINNER) return;
    const int row = idx / DINNER;
    const int d   = idx - row * DINNER;
    const int t   = row & (LTOK - 1);
    const float cur  = xz[(size_t)row * (2 * DINNER) + d];
    const float prev = (t == 0) ? 0.0f : xz[(size_t)(row - 1) * (2 * DINNER) + d];
    const float v    = prev * cw[2 * d] + cur * cw[2 * d + 1] + cb[d];
    xc[idx] = silu_f(v);
}

// ---------------------------------------------------------------------------
// Selective scan. One lane per (channel d, state n); 16 lanes per channel.
// block = 256 (16 channels x 16 states), grid = (DINNER/16, NBATCH).
// h_n <- exp(delta*A_dn)*h_n + delta*B_tn*xc ; y = sum_n h_n*C_tn (shfl tree)
// Fused epilogue: yg = (y + D_d*xc) * silu(z).
// ---------------------------------------------------------------------------
__global__ void scan_kernel(const float* __restrict__ delta,  // [NROWS,DINNER]
                            const float* __restrict__ dbc,    // [NROWS,64] (dt|B|C)
                            const float* __restrict__ xc,     // [NROWS,DINNER]
                            const float* __restrict__ xz,     // z at cols DINNER..
                            const float* __restrict__ A_log,  // [DINNER,16]
                            const float* __restrict__ Dv,     // [DINNER]
                            float* __restrict__ yg)           // [NROWS,DINNER]
{
    const int n  = threadIdx.x & 15;
    const int dl = threadIdx.x >> 4;
    const int d  = blockIdx.x * 16 + dl;
    const int b  = blockIdx.y;

    const float Ad = -expf(A_log[d * DSTATE + n]);
    const float Dd = Dv[d];
    float h = 0.0f;

    for (int t = 0; t < LTOK; ++t) {
        const int   row = b * LTOK + t;
        const float dlt = delta[(size_t)row * DINNER + d];
        const float xcv = xc[(size_t)row * DINNER + d];
        const float Bv  = dbc[row * 64 + DTRANK + n];
        const float Cv  = dbc[row * 64 + DTRANK + DSTATE + n];

        h = expf(dlt * Ad) * h + dlt * Bv * xcv;

        float yc = h * Cv;                       // reduce 16 states (wave32)
        yc += __shfl_xor(yc, 1, 32);
        yc += __shfl_xor(yc, 2, 32);
        yc += __shfl_xor(yc, 4, 32);
        yc += __shfl_xor(yc, 8, 32);

        if (n == 0) {
            const float zv = xz[(size_t)row * (2 * DINNER) + DINNER + d];
            yg[(size_t)row * DINNER + d] = (yc + Dd * xcv) * silu_f(zv);
        }
    }
}

// ---------------------------------------------------------------------------
// Sequence flip: dst[b,t,:] = src[b,L-1-t,:]
// ---------------------------------------------------------------------------
__global__ void flip_kernel(const float* __restrict__ src, float* __restrict__ dst)
{
    const int idx = blockIdx.x * blockDim.x + threadIdx.x;
    if (idx >= NROWS * DMODEL) return;
    const int row = idx / DMODEL;
    const int c   = idx - row * DMODEL;
    const int b   = row / LTOK, t = row - b * LTOK;
    dst[idx] = src[(size_t)(b * LTOK + (LTOK - 1 - t)) * DMODEL + c];
}

// ---------------------------------------------------------------------------
// out = LayerNorm(a + bsrc [+ csrc(optionally seq-flipped)])
// block = 256 (2 cols/thread over DMODEL=512), grid = NROWS.
// ---------------------------------------------------------------------------
__global__ void add_ln_kernel(const float* __restrict__ a,
                              const float* __restrict__ bsrc,
                              const float* __restrict__ csrc, int flip_c,
                              const float* __restrict__ g,
                              const float* __restrict__ beta,
                              float* __restrict__ out)
{
    __shared__ float red1[8], red2[8];
    const int row = blockIdx.x;
    const int c0  = threadIdx.x;              // 0..255
    const int b   = row / LTOK, t = row - b * LTOK;
    const int crow = flip_c ? (b * LTOK + (LTOK - 1 - t)) : row;

    float v0 = a[(size_t)row * DMODEL + c0]       + bsrc[(size_t)row * DMODEL + c0];
    float v1 = a[(size_t)row * DMODEL + c0 + 256] + bsrc[(size_t)row * DMODEL + c0 + 256];
    if (csrc) {
        v0 += csrc[(size_t)crow * DMODEL + c0];
        v1 += csrc[(size_t)crow * DMODEL + c0 + 256];
    }

    float s1 = v0 + v1;
    float s2 = v0 * v0 + v1 * v1;
#pragma unroll
    for (int off = 16; off > 0; off >>= 1) {
        s1 += __shfl_xor(s1, off, 32);
        s2 += __shfl_xor(s2, off, 32);
    }
    const int w = threadIdx.x >> 5;
    if ((threadIdx.x & 31) == 0) { red1[w] = s1; red2[w] = s2; }
    __syncthreads();
    if (threadIdx.x < 32) {
        float a1 = (threadIdx.x < 8) ? red1[threadIdx.x] : 0.0f;
        float a2 = (threadIdx.x < 8) ? red2[threadIdx.x] : 0.0f;
#pragma unroll
        for (int off = 4; off > 0; off >>= 1) {
            a1 += __shfl_xor(a1, off, 32);
            a2 += __shfl_xor(a2, off, 32);
        }
        if (threadIdx.x == 0) { red1[0] = a1; red2[0] = a2; }
    }
    __syncthreads();

    const float mu  = red1[0] * (1.0f / DMODEL);
    const float var = red2[0] * (1.0f / DMODEL) - mu * mu;
    const float rs  = rsqrtf(var + 1e-5f);
    out[(size_t)row * DMODEL + c0]       = (v0 - mu) * rs * g[c0]       + beta[c0];
    out[(size_t)row * DMODEL + c0 + 256] = (v1 - mu) * rs * g[c0 + 256] + beta[c0 + 256];
}

// ---------------------------------------------------------------------------
static inline void launch_gemm(const float* A, int lda, const float* W, int ldw,
                               const float* bias, float* O, int ldo,
                               int M, int N, int K, int act, hipStream_t s)
{
    dim3 grid(N / 64, M / 64), block(128);
    gemm_wmma_kernel<<<grid, block, 0, s>>>(A, lda, W, ldw, bias, O, ldo, K, act);
}

extern "C" void kernel_launch(void* const* d_in, const int* in_sizes, int n_in,
                              void* d_out, int out_size, void* d_ws, size_t ws_size,
                              hipStream_t stream)
{
    const float* x = (const float*)d_in[0];
    const float* m_in_w[2]   = {(const float*)d_in[1],  (const float*)d_in[10]};
    const float* m_conv_w[2] = {(const float*)d_in[2],  (const float*)d_in[11]};
    const float* m_conv_b[2] = {(const float*)d_in[3],  (const float*)d_in[12]};
    const float* m_xproj[2]  = {(const float*)d_in[4],  (const float*)d_in[13]};
    const float* m_dt_w[2]   = {(const float*)d_in[5],  (const float*)d_in[14]};
    const float* m_dt_b[2]   = {(const float*)d_in[6],  (const float*)d_in[15]};
    const float* m_A_log[2]  = {(const float*)d_in[7],  (const float*)d_in[16]};
    const float* m_Dv[2]     = {(const float*)d_in[8],  (const float*)d_in[17]};
    const float* m_out_w[2]  = {(const float*)d_in[9],  (const float*)d_in[18]};
    const float* pu_w = (const float*)d_in[19];
    const float* pu_b = (const float*)d_in[20];
    const float* pl_w = (const float*)d_in[21];
    const float* pl_b = (const float*)d_in[22];
    const float* ln_g = (const float*)d_in[23];
    const float* ln_b = (const float*)d_in[24];
    float* out = (float*)d_out;

    // Workspace layout (floats); m2 reuses m1's transient buffers, MLP aliases them.
    float* ws  = (float*)d_ws;
    size_t off = 0;
    float* xz    = ws + off; off += (size_t)NROWS * 2 * DINNER;  // 4.0M f
    float* xc    = ws + off; off += (size_t)NROWS * DINNER;      // 2.0M f
    float* dbc   = ws + off; off += (size_t)NROWS * 64;          // 128K f
    float* delta = ws + off; off += (size_t)NROWS * DINNER;      // 2.0M f
    float* yge   = ws + off; off += (size_t)NROWS * DINNER;      // 2.0M f
    float* y1    = ws + off; off += (size_t)NROWS * DMODEL;      // 1.0M f
    float* y2    = ws + off; off += (size_t)NROWS * DMODEL;      // 1.0M f
    float* y3    = ws + off; off += (size_t)NROWS * DMODEL;      // 1.0M f
    float* hbuf  = xz;    // MLP hidden aliases xz (mambas done by then)
    float* yp    = xc;    // MLP output aliases xc
    float* xflip = yge;   // flipped input aliases yge (consumed before scan writes)

    for (int m = 0; m < 2; ++m) {
        const float* inx = x;
        if (m == 1) {
            flip_kernel<<<(NROWS * DMODEL + 255) / 256, 256, 0, stream>>>(x, xflip);
            inx = xflip;
        }
        // xz = inx @ in_w^T                      [2048,512] x [2048,512]^T
        launch_gemm(inx, DMODEL, m_in_w[m], DMODEL, nullptr,
                    xz, 2 * DINNER, NROWS, 2 * DINNER, DMODEL, 0, stream);
        // xc = silu(causal depthwise conv(xi))
        conv_silu_kernel<<<(NROWS * DINNER + 255) / 256, 256, 0, stream>>>(
            xz, m_conv_w[m], m_conv_b[m], xc);
        // dbc = xc @ xproj^T                     [2048,1024] x [64,1024]^T
        launch_gemm(xc, DINNER, m_xproj[m], DINNER, nullptr,
                    dbc, 64, NROWS, 64, DINNER, 0, stream);
        // delta = softplus(dt @ dt_w^T + dt_b)   [2048,32] x [1024,32]^T
        launch_gemm(dbc, 64, m_dt_w[m], DTRANK, m_dt_b[m],
                    delta, DINNER, NROWS, DINNER, DTRANK, 2, stream);
        // selective scan + D*xc + silu(z) gate
        scan_kernel<<<dim3(DINNER / 16, NBATCH), 256, 0, stream>>>(
            delta, dbc, xc, xz, m_A_log[m], m_Dv[m], yge);
        // y = yg @ out_w^T                       [2048,1024] x [512,1024]^T
        launch_gemm(yge, DINNER, m_out_w[m], DINNER, nullptr,
                    (m == 0 ? y1 : y2), DMODEL, NROWS, DMODEL, DINNER, 0, stream);
    }

    // y3 = LN(x + y1 + flip(y2))
    add_ln_kernel<<<NROWS, 256, 0, stream>>>(x, y1, y2, 1, ln_g, ln_b, y3);
    // MLP: relu(y3 @ pu^T + pu_b) @ pl^T + pl_b
    launch_gemm(y3, DMODEL, pu_w, DMODEL, pu_b, hbuf, HIDDIM, NROWS, HIDDIM, DMODEL, 1, stream);
    launch_gemm(hbuf, HIDDIM, pl_w, HIDDIM, pl_b, yp, DMODEL, NROWS, DMODEL, HIDDIM, 0, stream);
    // out = LN(yp + y3)
    add_ln_kernel<<<NROWS, 256, 0, stream>>>(yp, y3, nullptr, 0, ln_g, ln_b, out);
}
